// HybridLayer_37134287241954
// MI455X (gfx1250) — matrix operand order repaired and checked by reading, not compiled, not run on hardware
//
#include <hip/hip_runtime.h>

// ---------------------------------------------------------------------------
// CDNA5 (gfx1250) hybrid layer: router + LN + GQA flash-attn + SSM scan + FFN
// - All GEMM-shaped math: v_wmma_f32_16x16x32_bf16 (f32 accumulate)
// - Row-major tile staging: global_load_async_to_lds_b128 (ASYNCcnt path)
// ---------------------------------------------------------------------------

typedef __attribute__((ext_vector_type(16))) __bf16 v16bf;
typedef __attribute__((ext_vector_type(8)))  float  v8f;

__device__ __forceinline__ unsigned short f2bf(float f) {
  unsigned int u = __float_as_uint(f);
  unsigned int r = u + 0x7fffu + ((u >> 16) & 1u);   // round-to-nearest-even
  return (unsigned short)(r >> 16);
}

// LDS byte address of a shared-memory pointer (flat LDS addr truncates to
// addr[31:0] per CDNA5 aperture rules)
__device__ __forceinline__ unsigned int lds_addr32(const void* p) {
  return (unsigned int)(unsigned long long)p;
}

// async copy of 16 bytes per lane, global -> LDS (per-lane addresses);
// INST_OFFSET is added on BOTH the global and LDS side per the ISA.
__device__ __forceinline__ void async_b128(unsigned int lds, const void* gaddr) {
  asm volatile("global_load_async_to_lds_b128 %0, %1, off"
               :: "v"(lds), "v"(gaddr) : "memory");
}
__device__ __forceinline__ void async_b128_off16(unsigned int lds, const void* gaddr) {
  asm volatile("global_load_async_to_lds_b128 %0, %1, off offset:16"
               :: "v"(lds), "v"(gaddr) : "memory");
}
__device__ __forceinline__ void async_b128_off32(unsigned int lds, const void* gaddr) {
  asm volatile("global_load_async_to_lds_b128 %0, %1, off offset:32"
               :: "v"(lds), "v"(gaddr) : "memory");
}
__device__ __forceinline__ void async_b128_off48(unsigned int lds, const void* gaddr) {
  asm volatile("global_load_async_to_lds_b128 %0, %1, off offset:48"
               :: "v"(lds), "v"(gaddr) : "memory");
}
__device__ __forceinline__ void wait_async0() {
  asm volatile("s_wait_asynccnt 0" ::: "memory");
}

// --- WMMA fragment loaders from LDS tiles with stride 72 ushorts (144B) ----
// A fragment 16x32 bf16: lane m=lane%16; half=lane/16; K = half*16 .. half*16+15 contiguous.
__device__ __forceinline__ v16bf ldsA_frag(const unsigned short (*S)[72], int m, int lane, int ks) {
  union { v16bf v; unsigned int u[8]; } f;
  const int half = lane >> 4;
  const int koff = ks * 32 + half * 16;
  const unsigned int* p0 = (const unsigned int*)&S[m][koff];
  const unsigned int* p1 = (const unsigned int*)&S[m][koff + 8];
#pragma unroll
  for (int i = 0; i < 4; ++i) { f.u[i] = p0[i]; f.u[4 + i] = p1[i]; }
  return f.v;
}
// B fragment 32x16 bf16 from a transposed tile Bt[n][k]: lane n=lane%16,
// K = half*16 .. half*16+15 contiguous -> one 32B contiguous LDS read.
__device__ __forceinline__ v16bf ldsB_frag(const unsigned short (*S)[72], int n, int lane, int ks) {
  union { v16bf v; unsigned int u[8]; } f;
  const int kb = ks * 32 + (lane >> 4) * 16;
  const unsigned int* p = (const unsigned int*)&S[n][kb];
#pragma unroll
  for (int i = 0; i < 8; ++i) f.u[i] = p[i];
  return f.v;
}

// ---------------------------------------------------------------------------
// Router scores: scores[b,t] = x[b,t,:] . Wg + bg
// ---------------------------------------------------------------------------
__global__ __launch_bounds__(256) void scores_kernel(const float* __restrict__ x,
                                                     const float* __restrict__ Wg,
                                                     const float* __restrict__ bg,
                                                     float* __restrict__ scores, int D) {
  __shared__ float red[256];
  const size_t token = blockIdx.x;
  const float* xp = x + token * D;
  float s = 0.f;
  for (int i = threadIdx.x; i < D; i += 256) s += xp[i] * Wg[i];
  red[threadIdx.x] = s; __syncthreads();
  for (int o = 128; o > 0; o >>= 1) {
    if (threadIdx.x < o) red[threadIdx.x] += red[threadIdx.x + o];
    __syncthreads();
  }
  if (threadIdx.x == 0) scores[token] = red[0] + bg[0];
}

// top-k by rank: token selected iff fewer than k tokens rank above it
// (score desc, index asc tie-break == jax top_k first-occurrence behavior)
__global__ void mask_kernel(const float* __restrict__ scores, float* __restrict__ mask,
                            int T, int k, int total) {
  int idx = blockIdx.x * blockDim.x + threadIdx.x;
  if (idx >= total) return;
  const int b = idx / T, t = idx % T;
  const float s = scores[idx];
  const float* row = scores + (size_t)b * T;
  int cnt = 0;
  for (int j = 0; j < T; ++j) {
    float sj = row[j];
    cnt += (sj > s) || (sj == s && j < t);
  }
  mask[idx] = (cnt < k) ? 1.f : 0.f;
}

// ---------------------------------------------------------------------------
// LayerNorm over D; writes f32 and (optionally) bf16 copy for WMMA inputs
// ---------------------------------------------------------------------------
__global__ __launch_bounds__(256) void ln_kernel(const float* __restrict__ x,
                                                 const float* __restrict__ w,
                                                 const float* __restrict__ bsh,
                                                 float* __restrict__ outF,
                                                 unsigned short* __restrict__ outH, int D) {
  __shared__ float red[256];
  __shared__ float mu_s, rstd_s;
  const size_t token = blockIdx.x;
  const float* xp = x + token * D;
  float s = 0.f;
  for (int i = threadIdx.x; i < D; i += 256) s += xp[i];
  red[threadIdx.x] = s; __syncthreads();
  for (int o = 128; o > 0; o >>= 1) { if (threadIdx.x < o) red[threadIdx.x] += red[threadIdx.x + o]; __syncthreads(); }
  if (threadIdx.x == 0) mu_s = red[0] / (float)D;
  __syncthreads();
  const float mu = mu_s;
  float v = 0.f;
  for (int i = threadIdx.x; i < D; i += 256) { float d = xp[i] - mu; v += d * d; }
  red[threadIdx.x] = v; __syncthreads();
  for (int o = 128; o > 0; o >>= 1) { if (threadIdx.x < o) red[threadIdx.x] += red[threadIdx.x + o]; __syncthreads(); }
  if (threadIdx.x == 0) rstd_s = rsqrtf(red[0] / (float)D + 1e-5f);
  __syncthreads();
  const float rstd = rstd_s;
  for (int i = threadIdx.x; i < D; i += 256) {
    float o = (xp[i] - mu) * rstd * w[i] + bsh[i];
    outF[token * D + i] = o;
    if (outH) outH[token * D + i] = f2bf(o);
  }
}

// f32 -> bf16 cast (weights / activations)
__global__ void cvt_bf16_kernel(const float* __restrict__ in, unsigned short* __restrict__ out, int n) {
  int i = blockIdx.x * blockDim.x + threadIdx.x;
  const int stride = gridDim.x * blockDim.x;
  for (; i < n; i += stride) out[i] = f2bf(in[i]);
}

// f32 transpose (for A -> A^T, coalesces the SSM inner product)
__global__ __launch_bounds__(256) void transpose_kernel(const float* __restrict__ in,
                                                        float* __restrict__ out, int D) {
  __shared__ float tile[32][33];
  const int bx = blockIdx.x * 32, by = blockIdx.y * 32;
  const int tx = threadIdx.x & 31, ty = threadIdx.x >> 5;
#pragma unroll
  for (int i = 0; i < 32; i += 8) tile[ty + i][tx] = in[(size_t)(by + ty + i) * D + bx + tx];
  __syncthreads();
#pragma unroll
  for (int i = 0; i < 32; i += 8) out[(size_t)(bx + ty + i) * D + by + tx] = tile[tx][ty + i];
}

// ---------------------------------------------------------------------------
// Tiled bf16 WMMA GEMM: C[M,N] = A[M,K] * B[K,N], f32 accumulate.
// Block: 256 threads = 8 waves, 64x64 output tile, K staged 64 at a time.
// A tile goes global->LDS via async-DMA (ASYNCcnt); B tile is transposed
// through VGPRs into Bt[n][k] so B fragments are contiguous LDS reads.
// mode 0: store f32 | 1: +bias, exact GELU | 2: +bias, +residual | 3: plain
// Store dtype selected by outH != nullptr (bf16) else outF (f32).
// ---------------------------------------------------------------------------
__global__ __launch_bounds__(256) void gemm_bf16_kernel(
    const unsigned short* __restrict__ A, const unsigned short* __restrict__ B,
    float* __restrict__ outF, unsigned short* __restrict__ outH,
    const float* __restrict__ bias, const float* __restrict__ resid,
    int M, int N, int K, int mode) {
  __shared__ alignas(16) unsigned short As[64][72];   // A tile, row-major
  __shared__ alignas(16) unsigned short Bts[64][72];  // B tile, transposed [n][k]

  const int tid = threadIdx.x;
  const int lane = tid & 31;
  const int w = tid >> 5;      // 0..7
  const int wm = w & 3;        // 4 M sub-tiles of 16
  const int wn = w >> 2;       // 2 N halves of 32
  const int bn = blockIdx.x * 64;
  const int bm = blockIdx.y * 64;

  v8f acc[2] = { {0,0,0,0,0,0,0,0}, {0,0,0,0,0,0,0,0} };

  for (int k0 = 0; k0 < K; k0 += 64) {
    __syncthreads();  // previous tile fully consumed before overwrite
    {  // stage A 64x64: 32B per thread, direct global->LDS (async)
      const int r = tid >> 2;
      const int c = (tid & 3) * 16;
      const unsigned short* gp = A + (size_t)(bm + r) * K + k0 + c;
      const unsigned int lo = lds_addr32(&As[r][c]);
      async_b128(lo, gp);
      async_b128_off16(lo, gp);   // offset applies to both global and LDS side
      if (k0 + 64 < K) __builtin_prefetch(gp + 64, 0, 1);  // global_prefetch_b8
    }
    {  // stage B 64x64 transposed into Bts[n][k]
      const int r = tid >> 2;              // k row within tile
      const int c = (tid & 3) * 16;        // n chunk
      const unsigned short* gp = B + (size_t)(k0 + r) * N + bn + c;
      unsigned short tmp[16];
      *(uint4*)&tmp[0] = *(const uint4*)gp;
      *(uint4*)&tmp[8] = *(const uint4*)(gp + 8);
#pragma unroll
      for (int i = 0; i < 16; ++i) Bts[c + i][r] = tmp[i];
    }
    wait_async0();    // this wave's async LDS writes are visible
    __syncthreads();  // everyone's staging is visible

#pragma unroll
    for (int ks = 0; ks < 2; ++ks) {
      const v16bf fa = ldsA_frag(As, (lane & 15) + wm * 16, lane, ks);
#pragma unroll
      for (int nt = 0; nt < 2; ++nt) {
        const v16bf fb = ldsB_frag(Bts, (lane & 15) + wn * 32 + nt * 16, lane, ks);
        acc[nt] = __builtin_amdgcn_wmma_f32_16x16x32_bf16(
            false, fa, false, fb, (short)0, acc[nt], false, false);
      }
    }
  }

  // epilogue (C layout: row = half*8 + r, col = lane%16)
#pragma unroll
  for (int nt = 0; nt < 2; ++nt) {
    const int col = bn + wn * 32 + nt * 16 + (lane & 15);
#pragma unroll
    for (int r = 0; r < 8; ++r) {
      const int row = bm + wm * 16 + (lane >> 4) * 8 + r;
      float v = acc[nt][r];
      if (mode == 1) { v += bias[col]; v = 0.5f * v * (1.f + erff(v * 0.70710678118654752f)); }
      else if (mode == 2) { v += bias[col]; v += resid[(size_t)row * N + col]; }
      const size_t o = (size_t)row * N + col;
      if (outH) outH[o] = f2bf(v);
      else      outF[o] = v;
    }
  }
}

// ---------------------------------------------------------------------------
// Flash attention, n_kv_heads = 1 (K/V shared across 16 heads), hd = 64.
// Block = 128 threads (4 waves), 64 queries per block, key tiles of 64.
// WMMA for S = Q K^T and O += P V; online softmax state in LDS.
// Q and K tiles staged via async global->LDS; V transposed through VGPRs.
// ---------------------------------------------------------------------------
__global__ __launch_bounds__(128) void flash_attn_kernel(
    const unsigned short* __restrict__ Qb,  // [B*T, D] bf16 (head h at cols h*64)
    const unsigned short* __restrict__ Kb,  // [B*T, 64] bf16
    const unsigned short* __restrict__ Vb,  // [B*T, 64] bf16
    unsigned short* __restrict__ Ob,        // [B*T, D] bf16
    int T, int D) {
  const int hd = 64;
  const int qb0 = blockIdx.x * 64;
  const int h = blockIdx.y;
  const int b = blockIdx.z;

  __shared__ alignas(16) unsigned short Qs[64][72];
  __shared__ alignas(16) unsigned short Ks[64][72];   // Bt for S: Ks[key][hdk]
  __shared__ alignas(16) unsigned short Vts[64][72];  // Bt for O: Vts[hd][key]
  __shared__ alignas(16) unsigned short Ps[64][72];   // P, A-operand of PV
  __shared__ float Ss[64][68];
  __shared__ float m_s[64], l_s[64], al_s[64];

  const int tid = threadIdx.x;
  const int lane = tid & 31;
  const int w = tid >> 5;  // wave = M sub-tile (rows w*16..w*16+15)

  {  // stage Q tile once: 64B per thread, async global->LDS
    const int r = tid >> 1;
    const int c0 = (tid & 1) * 32;
    const unsigned short* gp = Qb + (size_t)(b * T + qb0 + r) * D + h * hd + c0;
    const unsigned int lo = lds_addr32(&Qs[r][c0]);
    async_b128(lo, gp);
    async_b128_off16(lo, gp);
    async_b128_off32(lo, gp);
    async_b128_off48(lo, gp);
  }
  if (tid < 64) { m_s[tid] = -INFINITY; l_s[tid] = 0.f; }

  v8f acc[4] = { {0,0,0,0,0,0,0,0}, {0,0,0,0,0,0,0,0},
                 {0,0,0,0,0,0,0,0}, {0,0,0,0,0,0,0,0} };

  const int nKT = T / 64;
  for (int kt = 0; kt < nKT; ++kt) {
    __syncthreads();  // prior tile consumed
    {  // stage K (row-major, async) and V (transposed through VGPRs)
      const int r = tid >> 1;
      const int c0 = (tid & 1) * 32;
      const unsigned short* gpk = Kb + (size_t)(b * T + kt * 64 + r) * hd + c0;
      const unsigned int lo = lds_addr32(&Ks[r][c0]);
      async_b128(lo, gpk);
      async_b128_off16(lo, gpk);
      async_b128_off32(lo, gpk);
      async_b128_off48(lo, gpk);
      const unsigned short* gpv = Vb + (size_t)(b * T + kt * 64 + r) * hd + c0;
      unsigned short tmp[32];
#pragma unroll
      for (int i = 0; i < 4; ++i)
        *(uint4*)&tmp[i * 8] = *(const uint4*)(gpv + i * 8);
#pragma unroll
      for (int i = 0; i < 32; ++i) Vts[c0 + i][r] = tmp[i];
    }
    wait_async0();   // covers this iteration's K loads (and the Q load on kt==0)
    __syncthreads();

    // S = Q K^T (16x64 per wave)
    v8f sacc[4] = { {0,0,0,0,0,0,0,0}, {0,0,0,0,0,0,0,0},
                    {0,0,0,0,0,0,0,0}, {0,0,0,0,0,0,0,0} };
#pragma unroll
    for (int ks = 0; ks < 2; ++ks) {
      const v16bf fa = ldsA_frag(Qs, (lane & 15) + w * 16, lane, ks);
#pragma unroll
      for (int nt = 0; nt < 4; ++nt) {
        const v16bf fb = ldsB_frag(Ks, (lane & 15) + nt * 16, lane, ks);
        sacc[nt] = __builtin_amdgcn_wmma_f32_16x16x32_bf16(
            false, fa, false, fb, (short)0, sacc[nt], false, false);
      }
    }
#pragma unroll
    for (int nt = 0; nt < 4; ++nt)
#pragma unroll
      for (int r = 0; r < 8; ++r)
        Ss[w * 16 + (lane >> 4) * 8 + r][nt * 16 + (lane & 15)] = sacc[nt][r] * 0.125f;
    __syncthreads();

    // online softmax: one thread per query row
    if (tid < 64) {
      const int r = tid;
      const float mo = m_s[r];
      float mx = mo;
#pragma unroll 8
      for (int j = 0; j < 64; ++j) mx = fmaxf(mx, Ss[r][j]);
      const float alpha = expf(mo - mx);
      float sum = 0.f;
#pragma unroll 8
      for (int j = 0; j < 64; ++j) {
        float p = expf(Ss[r][j] - mx);
        sum += p;
        Ps[r][j] = f2bf(p);
      }
      l_s[r] = l_s[r] * alpha + sum;
      m_s[r] = mx;
      al_s[r] = alpha;
    }
    __syncthreads();

    // rescale running O and accumulate P * V
#pragma unroll
    for (int nt = 0; nt < 4; ++nt)
#pragma unroll
      for (int r = 0; r < 8; ++r)
        acc[nt][r] *= al_s[w * 16 + (lane >> 4) * 8 + r];
#pragma unroll
    for (int ks = 0; ks < 2; ++ks) {
      const v16bf fa = ldsA_frag(Ps, (lane & 15) + w * 16, lane, ks);
#pragma unroll
      for (int nt = 0; nt < 4; ++nt) {
        const v16bf fb = ldsB_frag(Vts, (lane & 15) + nt * 16, lane, ks);
        acc[nt] = __builtin_amdgcn_wmma_f32_16x16x32_bf16(
            false, fa, false, fb, (short)0, acc[nt], false, false);
      }
    }
  }

  // epilogue: O / l, write bf16 at the head's column block
#pragma unroll
  for (int nt = 0; nt < 4; ++nt) {
    const int col = nt * 16 + (lane & 15);
#pragma unroll
    for (int r = 0; r < 8; ++r) {
      const int row = w * 16 + (lane >> 4) * 8 + r;
      const float v = acc[nt][r] / l_s[row];
      Ob[(size_t)(b * T + qb0 + row) * D + h * hd + col] = f2bf(v);
    }
  }
}

// ---------------------------------------------------------------------------
// Sequential SSM scan. Batches independent -> one workgroup (1024 thr) per b,
// hidden state h[1024] resident in LDS; A^T/Wc read column-coalesced from L2.
//   h = tanh(h @ A^T + xb_t);  y = h @ Wc + Dp * xn_t;  deep += y
// ---------------------------------------------------------------------------
__global__ __launch_bounds__(1024) void ssm_kernel(
    const float* __restrict__ xb, const float* __restrict__ xn,
    const float* __restrict__ At, const float* __restrict__ Wc,
    const float* __restrict__ Dp, float* __restrict__ deep, int T, int D) {
  __shared__ float h[1024];
  const int b = blockIdx.x;
  const int d = threadIdx.x;
  h[d] = 0.f;
  const float dp = Dp[d];
  __syncthreads();
  for (int t = 0; t < T; ++t) {
    float s = 0.f;
    for (int k = 0; k < D; ++k) s += h[k] * At[(size_t)k * D + d];
    const float hn = tanhf(s + xb[((size_t)b * T + t) * D + d]);
    __syncthreads();
    h[d] = hn;
    __syncthreads();
    float y = 0.f;
    for (int k = 0; k < D; ++k) y += h[k] * Wc[(size_t)k * D + d];
    y += dp * xn[((size_t)b * T + t) * D + d];
    deep[((size_t)b * T + t) * D + d] += y;
  }
}

// x2 = residual + deep*m + residual*skip_gate*(1-m)
__global__ void fuse_kernel(const float* __restrict__ x, const float* __restrict__ deep,
                            const float* __restrict__ mask, const float* __restrict__ sg,
                            float* __restrict__ x2, int D, int n) {
  const int i = blockIdx.x * blockDim.x + threadIdx.x;
  if (i >= n) return;
  const float m = mask[i / D];
  const float r = x[i];
  x2[i] = r + deep[i] * m + r * sg[0] * (1.f - m);
}

// ---------------------------------------------------------------------------
extern "C" void kernel_launch(void* const* d_in, const int* in_sizes, int n_in,
                              void* d_out, int out_size, void* d_ws, size_t ws_size,
                              hipStream_t stream) {
  (void)in_sizes; (void)n_in; (void)out_size; (void)ws_size;
  const int B = 2, T = 2048, D = 1024, H = 16, hd = 64;
  const size_t BT = (size_t)B * T;     // 4096
  const size_t BTD = BT * D;           // 4,194,304

  const float* x    = (const float*)d_in[0];
  const float* Wg   = (const float*)d_in[1];
  const float* bg   = (const float*)d_in[2];
  const float* Wq   = (const float*)d_in[3];
  const float* Wk   = (const float*)d_in[4];
  const float* Wv   = (const float*)d_in[5];
  const float* Wo   = (const float*)d_in[6];
  const float* Amat = (const float*)d_in[7];
  const float* Wb   = (const float*)d_in[8];
  const float* Wc   = (const float*)d_in[9];
  const float* Dp   = (const float*)d_in[10];
  const float* W1   = (const float*)d_in[11];
  const float* b1   = (const float*)d_in[12];
  const float* W2   = (const float*)d_in[13];
  const float* b2   = (const float*)d_in[14];
  const float* ln1w = (const float*)d_in[15];
  const float* ln1b = (const float*)d_in[16];
  const float* ln2w = (const float*)d_in[17];
  const float* ln2b = (const float*)d_in[18];
  const float* sg   = (const float*)d_in[19];

  float* out    = (float*)d_out;     // [B,T,D] then scores [B,T]
  float* scores = out + BTD;

  char* ws = (char*)d_ws;
  size_t off = 0;
  auto alloc = [&](size_t bytes) -> void* {
    off = (off + 255) & ~(size_t)255;
    void* p = (void*)(ws + off);
    off += bytes;
    return p;
  };

  float* xn   = (float*)alloc(BTD * 4);
  float* x2   = (float*)alloc(BTD * 4);
  float* deep = (float*)alloc(BTD * 4);
  float* xb   = (float*)alloc(BTD * 4);
  float* xn2  = (float*)alloc(BTD * 4);
  float* At   = (float*)alloc((size_t)D * D * 4);
  float* maskb = (float*)alloc(BT * 4);
  unsigned short* xn_bf   = (unsigned short*)alloc(BTD * 2);
  unsigned short* xn2_bf  = (unsigned short*)alloc(BTD * 2);
  unsigned short* q_bf    = (unsigned short*)alloc(BTD * 2);
  unsigned short* k_bf    = (unsigned short*)alloc(BT * hd * 2);
  unsigned short* v_bf    = (unsigned short*)alloc(BT * hd * 2);
  unsigned short* attn_bf = (unsigned short*)alloc(BTD * 2);
  unsigned short* ffn1_bf = (unsigned short*)alloc(BT * (size_t)(4 * D) * 2);
  unsigned short* Wq_bf = (unsigned short*)alloc((size_t)D * D * 2);
  unsigned short* Wk_bf = (unsigned short*)alloc((size_t)D * hd * 2);
  unsigned short* Wv_bf = (unsigned short*)alloc((size_t)D * hd * 2);
  unsigned short* Wo_bf = (unsigned short*)alloc((size_t)D * D * 2);
  unsigned short* Wb_bf = (unsigned short*)alloc((size_t)D * D * 2);
  unsigned short* W1_bf = (unsigned short*)alloc((size_t)D * 4 * D * 2);
  unsigned short* W2_bf = (unsigned short*)alloc((size_t)4 * D * D * 2);

  auto gemm = [&](const unsigned short* Am, const unsigned short* Bm,
                  float* oF, unsigned short* oH, const float* bias, const float* resid,
                  int M, int N, int K, int mode) {
    dim3 g(N / 64, M / 64);
    gemm_bf16_kernel<<<g, 256, 0, stream>>>(Am, Bm, oF, oH, bias, resid, M, N, K, mode);
  };
  auto cvt = [&](const float* in, unsigned short* o, int n) {
    cvt_bf16_kernel<<<2048, 256, 0, stream>>>(in, o, n);
  };

  // 1) router scores (directly into d_out tail) + top-k mask
  scores_kernel<<<(int)BT, 256, 0, stream>>>(x, Wg, bg, scores, D);
  mask_kernel<<<((int)BT + 255) / 256, 256, 0, stream>>>(scores, maskb, T, T / 4, (int)BT);

  // 2) LN1 (f32 + bf16)
  ln_kernel<<<(int)BT, 256, 0, stream>>>(x, ln1w, ln1b, xn, xn_bf, D);

  // 3) weight casts + A^T
  cvt(Wq, Wq_bf, D * D);   cvt(Wk, Wk_bf, D * hd);  cvt(Wv, Wv_bf, D * hd);
  cvt(Wo, Wo_bf, D * D);   cvt(Wb, Wb_bf, D * D);
  cvt(W1, W1_bf, D * 4 * D);  cvt(W2, W2_bf, 4 * D * D);
  transpose_kernel<<<dim3(D / 32, D / 32), 256, 0, stream>>>(Amat, At, D);

  // 4) Q/K/V projections (bf16 WMMA, bf16 out for attention)
  gemm(xn_bf, Wq_bf, nullptr, q_bf, nullptr, nullptr, (int)BT, D, D, 3);
  gemm(xn_bf, Wk_bf, nullptr, k_bf, nullptr, nullptr, (int)BT, hd, D, 3);
  gemm(xn_bf, Wv_bf, nullptr, v_bf, nullptr, nullptr, (int)BT, hd, D, 3);

  // 5) flash attention (scores never materialized), then x Wo -> deep
  flash_attn_kernel<<<dim3(T / 64, H, B), 128, 0, stream>>>(q_bf, k_bf, v_bf, attn_bf, T, D);
  gemm(attn_bf, Wo_bf, deep, nullptr, nullptr, nullptr, (int)BT, D, D, 0);

  // 6) SSM: precompute xb = xn Wb (parallel), then sequential scan adds into deep
  gemm(xn_bf, Wb_bf, xb, nullptr, nullptr, nullptr, (int)BT, D, D, 0);
  ssm_kernel<<<B, 1024, 0, stream>>>(xb, xn, At, Wc, Dp, deep, T, D);

  // 7) routing mix -> x2 ; LN2
  fuse_kernel<<<((int)BTD + 255) / 256, 256, 0, stream>>>(x, deep, maskb, sg, x2, D, (int)BTD);
  ln_kernel<<<(int)BT, 256, 0, stream>>>(x2, ln2w, ln2b, xn2, xn2_bf, D);

  // 8) FFN: W1 + bias + exact GELU (bf16 out), then W2 + bias + residual -> d_out
  gemm(xn2_bf, W1_bf, nullptr, ffn1_bf, b1, nullptr, (int)BT, 4 * D, D, 1);
  gemm(ffn1_bf, W2_bf, out, nullptr, b2, x2, (int)BT, D, 4 * D, 2);
}
